// MinGRU_47794396070431
// MI455X (gfx1250) — compile-verified, matrix operand order checked by reading, not run on hardware
//
#include <hip/hip_runtime.h>
#include <hip/hip_bf16.h>
#include <math.h>

// ---------------------------------------------------------------------------
// minGRU for MI455X (gfx1250): bf16 WMMA dual-GEMM with async global->LDS
// double-buffered staging and fused log-space epilogue, followed by a
// 3-pass chunked associative scan.  B=4, T=4096, D=H=1024.
// ---------------------------------------------------------------------------

typedef __bf16 bf16_t;
typedef __attribute__((ext_vector_type(16))) __bf16 v16bf;
typedef __attribute__((ext_vector_type(8)))  __bf16 v8bf;
typedef __attribute__((ext_vector_type(4)))  __bf16 v4bf;
typedef __attribute__((ext_vector_type(8)))  float  v8f;
typedef __attribute__((ext_vector_type(4)))  float  v4f;

// ---------------- math helpers (numerically stable, log-space) -------------

__device__ __forceinline__ float softplusf(float x) {
  // log(1 + exp(x)) = max(x,0) + log1p(exp(-|x|))
  return fmaxf(x, 0.0f) + log1pf(expf(-fabsf(x)));
}

__device__ __forceinline__ float log_gf(float x) {
  // log g(x): x>=0 -> log(x + 0.5), else log sigmoid(x) = -softplus(-x)
  return (x >= 0.0f) ? logf(x + 0.5f) : -softplusf(-x);
}

__device__ __forceinline__ float logaddexpf(float a, float b) {
  float m = fmaxf(a, b);
  if (m == -INFINITY) return -INFINITY;
  return m + log1pf(expf(-fabsf(a - b)));
}

// ---------------- CDNA5 async global->LDS copy (ASYNCcnt path) -------------

__device__ __forceinline__ void async_copy_b128(bf16_t* lds_dst,
                                                const bf16_t* gsrc) {
  // Generic pointer to LDS: low 32 bits are the LDS byte offset.
  unsigned lds_off = (unsigned)(uintptr_t)lds_dst;
  unsigned long long gaddr = (unsigned long long)(uintptr_t)gsrc;
  asm volatile("global_load_async_to_lds_b128 %0, %1, off"
               :
               : "v"(lds_off), "v"(gaddr)
               : "memory");
}

__device__ __forceinline__ void wait_asynccnt0() {
  asm volatile("s_wait_asynccnt 0x0" ::: "memory");
}

// ---------------- f32 -> bf16 conversion (vectorized 4-wide) ---------------

__global__ __launch_bounds__(256) void cvt_f32_bf16(
    const float* __restrict__ src, bf16_t* __restrict__ dst, long n4) {
  long i = (long)blockIdx.x * blockDim.x + threadIdx.x;
  if (i >= n4) return;
  v4f x = *(const v4f*)(src + i * 4);
  v4bf y;
  y.x = (__bf16)x.x; y.y = (__bf16)x.y; y.z = (__bf16)x.z; y.w = (__bf16)x.w;
  *(v4bf*)(dst + i * 4) = y;
}

// ---------------- dual GEMM + fused log-space epilogue ---------------------
// Block: 128 threads = 4 waves.  Block tile 64(M) x 64(N); each wave owns a
// 16x64 strip for BOTH weight matrices (8 f32 accumulators of 8 VGPRs).
// K stepped by 32; A/Wz/Wh 64x32 tiles staged through LDS (double buffered)
// via global_load_async_to_lds_b128; fragments read back with ds_load_b128.

#define GEMM_KSTEP 32
#define LDS_ROWP   40                 // padded row stride (bf16 elems) = 80 B
#define TILE_ELEMS (64 * LDS_ROWP)    // one 64x32 tile (padded)
#define STAGE_ELEMS (3 * TILE_ELEMS)  // A, Wz, Wh tiles per stage

__global__ __launch_bounds__(128) void mingru_dual_gemm_wmma(
    const bf16_t* __restrict__ xb,   // [M, K] row-major bf16
    const bf16_t* __restrict__ wzb,  // [N, K] row-major bf16
    const bf16_t* __restrict__ whb,  // [N, K] row-major bf16
    const float*  __restrict__ bz,
    const float*  __restrict__ bh,
    float* __restrict__ lc,          // [M, N]
    float* __restrict__ lv,          // [M, N]
    int M, int K, int N) {
  __shared__ bf16_t smem[2 * STAGE_ELEMS];

  const int tid    = threadIdx.x;
  const int wave   = tid >> 5;
  const int lane   = tid & 31;
  const int laneN  = lane & 15;  // row/col within 16
  const int laneHi = lane >> 4;  // 0: K 0..15, 1: K 16..31
  const int m0b = blockIdx.x * 64;
  const int n0b = blockIdx.y * 64;

  // Cooperative load mapping: each 64x32 tile = 256 chunks of 16 B
  // (chunk c: row = c>>2, element offset = (c&3)*8).  128 threads take
  // chunks tid and tid+128 of every tile.
  const int c0 = tid,        c1 = tid + 128;
  const int r0 = c0 >> 2,    r1 = c1 >> 2;
  const int e0 = (c0 & 3) * 8, e1 = (c1 & 3) * 8;

  auto issue_stage = [&](int k0, int stage) {
    bf16_t* sb = smem + stage * STAGE_ELEMS;
    // A tile (rows m0b..m0b+63, cols k0..k0+31)
    async_copy_b128(sb + 0 * TILE_ELEMS + r0 * LDS_ROWP + e0,
                    xb + (size_t)(m0b + r0) * K + k0 + e0);
    async_copy_b128(sb + 0 * TILE_ELEMS + r1 * LDS_ROWP + e1,
                    xb + (size_t)(m0b + r1) * K + k0 + e1);
    // Wz tile (rows n0b..n0b+63)
    async_copy_b128(sb + 1 * TILE_ELEMS + r0 * LDS_ROWP + e0,
                    wzb + (size_t)(n0b + r0) * K + k0 + e0);
    async_copy_b128(sb + 1 * TILE_ELEMS + r1 * LDS_ROWP + e1,
                    wzb + (size_t)(n0b + r1) * K + k0 + e1);
    // Wh tile
    async_copy_b128(sb + 2 * TILE_ELEMS + r0 * LDS_ROWP + e0,
                    whb + (size_t)(n0b + r0) * K + k0 + e0);
    async_copy_b128(sb + 2 * TILE_ELEMS + r1 * LDS_ROWP + e1,
                    whb + (size_t)(n0b + r1) * K + k0 + e1);
  };

  // Fragment loader: 16 contiguous bf16 from LDS as two 16B-aligned halves.
  auto load_frag = [&](const bf16_t* p) -> v16bf {
    union { v8bf h[2]; v16bf f; } u;
    u.h[0] = *(const v8bf*)(p);
    u.h[1] = *(const v8bf*)(p + 8);
    return u.f;
  };

  v8f accK[4], accP[4];
#pragma unroll
  for (int i = 0; i < 4; ++i) {
    accK[i] = (v8f)(0.0f);
    accP[i] = (v8f)(0.0f);
  }

  issue_stage(0, 0);

  for (int k0 = 0; k0 < K; k0 += GEMM_KSTEP) {
    const int cur = (k0 >> 5) & 1;
    wait_asynccnt0();     // this stage's async loads (issued last iter) done
    __syncthreads();      // visible to all waves; prev stage reads retired
    if (k0 + GEMM_KSTEP < K) issue_stage(k0 + GEMM_KSTEP, cur ^ 1);

    const bf16_t* sb = smem + cur * STAGE_ELEMS;
    const int arow = wave * 16 + laneN;
    v16bf afrag =
        load_frag(sb + 0 * TILE_ELEMS + arow * LDS_ROWP + laneHi * 16);
#pragma unroll
    for (int nt = 0; nt < 4; ++nt) {
      const int brow = nt * 16 + laneN;
      v16bf bzf =
          load_frag(sb + 1 * TILE_ELEMS + brow * LDS_ROWP + laneHi * 16);
      accK[nt] = __builtin_amdgcn_wmma_f32_16x16x32_bf16(
          false, afrag, false, bzf, (short)0, accK[nt], false, false);
      v16bf bhf =
          load_frag(sb + 2 * TILE_ELEMS + brow * LDS_ROWP + laneHi * 16);
      accP[nt] = __builtin_amdgcn_wmma_f32_16x16x32_bf16(
          false, afrag, false, bhf, (short)0, accP[nt], false, false);
    }
  }

  // Epilogue: C/D layout -> VGPR r holds row m0 + r + 8*laneHi, col n0+laneN.
  const int m0 = m0b + wave * 16;
#pragma unroll
  for (int nt = 0; nt < 4; ++nt) {
    const int n = n0b + nt * 16 + laneN;
    const float bzv = bz[n];
    const float bhv = bh[n];
#pragma unroll
    for (int r = 0; r < 8; ++r) {
      const int m = m0 + r + 8 * laneHi;
      const float kk = accK[nt][r] + bzv;
      const float pp = accP[nt][r] + bhv;
      const size_t idx = (size_t)m * N + n;
      lc[idx] = -softplusf(kk);
      lv[idx] = -softplusf(-kk) + log_gf(pp);
    }
  }
}

// ---------------- pass 1: per-chunk scan aggregates ------------------------
// grid: (B, H/64, NC); block: 64 threads (one h-channel each).
// Aggregate (A,V) of a chunk satisfies  s_out = logaddexp(A + s_in, V).

__global__ __launch_bounds__(64) void scan_chunk_agg(
    const float* __restrict__ lc, const float* __restrict__ lv,
    float* __restrict__ Aagg, float* __restrict__ Vagg,
    int T, int H, int TC) {
  const int b = blockIdx.x;
  const int h = blockIdx.y * 64 + threadIdx.x;
  const int c = blockIdx.z;
  const size_t base = ((size_t)b * T + (size_t)c * TC) * H + h;

  float a = lc[base];
  float v = lv[base];
  for (int t = 1; t < TC; ++t) {
    const size_t idx = base + (size_t)t * H;
    const float ea = lc[idx];
    const float ev = lv[idx];
    a += ea;
    v = logaddexpf(ea + v, ev);
  }
  const size_t o = ((size_t)c * gridDim.x + b) * H + h;  // [NC][B][H]
  Aagg[o] = a;
  Vagg[o] = v;
}

// ---------------- pass 2: scan over chunk aggregates -----------------------

__global__ __launch_bounds__(256) void scan_aggregate(
    const float* __restrict__ Aagg, const float* __restrict__ Vagg,
    const float* __restrict__ h0, float* __restrict__ sin_buf,
    int BH, int NC) {
  const int i = blockIdx.x * blockDim.x + threadIdx.x;
  if (i >= BH) return;
  float s = log_gf(h0[i]);
  for (int c = 0; c < NC; ++c) {
    const size_t o = (size_t)c * BH + i;
    sin_buf[o] = s;
    s = logaddexpf(Aagg[o] + s, Vagg[o]);
  }
}

// ---------------- pass 3: replay chunks, write h = exp(log_h) --------------

__global__ __launch_bounds__(64) void scan_apply(
    const float* __restrict__ lc, const float* __restrict__ lv,
    const float* __restrict__ sin_buf, float* __restrict__ out,
    int T, int H, int TC, int B) {
  const int b = blockIdx.x;
  const int h = blockIdx.y * 64 + threadIdx.x;
  const int c = blockIdx.z;
  const int BH = B * H;

  float s = sin_buf[(size_t)c * BH + b * H + h];
  const size_t base = ((size_t)b * T + (size_t)c * TC) * H + h;
  for (int t = 0; t < TC; ++t) {
    const size_t idx = base + (size_t)t * H;
    s = logaddexpf(lc[idx] + s, lv[idx]);
    out[idx] = expf(s);
  }
}

// ---------------------------------------------------------------------------

extern "C" void kernel_launch(void* const* d_in, const int* in_sizes, int n_in,
                              void* d_out, int out_size, void* d_ws, size_t ws_size,
                              hipStream_t stream) {
  (void)n_in; (void)out_size; (void)ws_size;

  const float* x  = (const float*)d_in[0];   // [B, T, D]
  const float* h0 = (const float*)d_in[1];   // [B, H]
  const float* Wz = (const float*)d_in[2];   // [H, D]
  const float* bz = (const float*)d_in[3];   // [H]
  const float* Wh = (const float*)d_in[4];   // [H, D]
  const float* bh = (const float*)d_in[5];   // [H]
  float* out = (float*)d_out;                // [B, T, H]

  const int H = in_sizes[3];                 // 1024
  const int D = in_sizes[2] / H;             // 1024
  const int B = in_sizes[1] / H;             // 4
  const int T = in_sizes[0] / (B * D);       // 4096
  const int M = B * T;                       // 16384 rows of x
  const int NC = 64;                         // chunks along T
  const int TC = T / NC;                     // 64 steps per chunk
  const int BH = B * H;

  // ---- workspace carve-out (all sizes are multiples of 256B) ----
  char* ws = (char*)d_ws;
  bf16_t* xb   = (bf16_t*)ws;                 ws += (size_t)M * D * sizeof(bf16_t);
  bf16_t* wzb  = (bf16_t*)ws;                 ws += (size_t)H * D * sizeof(bf16_t);
  bf16_t* whb  = (bf16_t*)ws;                 ws += (size_t)H * D * sizeof(bf16_t);
  float*  lc   = (float*)ws;                  ws += (size_t)M * H * sizeof(float);
  float*  lv   = (float*)ws;                  ws += (size_t)M * H * sizeof(float);
  float*  Aagg = (float*)ws;                  ws += (size_t)NC * BH * sizeof(float);
  float*  Vagg = (float*)ws;                  ws += (size_t)NC * BH * sizeof(float);
  float*  sinb = (float*)ws;                  ws += (size_t)NC * BH * sizeof(float);

  // 1) convert activations + weights to bf16
  {
    long n4 = (long)M * D / 4;
    cvt_f32_bf16<<<(n4 + 255) / 256, 256, 0, stream>>>(x, xb, n4);
    long w4 = (long)H * D / 4;
    cvt_f32_bf16<<<(w4 + 255) / 256, 256, 0, stream>>>(Wz, wzb, w4);
    cvt_f32_bf16<<<(w4 + 255) / 256, 256, 0, stream>>>(Wh, whb, w4);
  }

  // 2) dual GEMM with fused log-space epilogue (WMMA + async LDS staging)
  {
    dim3 grid(M / 64, H / 64);
    mingru_dual_gemm_wmma<<<grid, 128, 0, stream>>>(
        xb, wzb, whb, bz, bh, lc, lv, M, D, H);
  }

  // 3) chunked associative scan
  {
    dim3 g1(B, H / 64, NC);
    scan_chunk_agg<<<g1, 64, 0, stream>>>(lc, lv, Aagg, Vagg, T, H, TC);

    scan_aggregate<<<(BH + 255) / 256, 256, 0, stream>>>(
        Aagg, Vagg, h0, sinb, BH, NC);

    dim3 g3(B, H / 64, NC);
    scan_apply<<<g3, 64, 0, stream>>>(lc, lv, sinb, out, T, H, TC, B);
  }
}
